// scLGF_64793876627463
// MI455X (gfx1250) — matrix operand order repaired and checked by reading, not compiled, not run on hardware
//
#include <hip/hip_runtime.h>
#include <hip/hip_bf16.h>
#include <stdint.h>

// ---------------------------------------------------------------------------
// CDNA5 (gfx1250) WMMA types
// ---------------------------------------------------------------------------
typedef __attribute__((ext_vector_type(16))) __bf16 v16bf;
typedef __attribute__((ext_vector_type(2)))  __bf16 v2bf;
typedef __attribute__((ext_vector_type(2)))  float  v2f;
typedef __attribute__((ext_vector_type(8)))  float  v8f;

union Frag {
    uint4  q[2];
    v16bf  v;
};

// native hardware bf16 conversion (v_cvt_*_bf16_f32 on gfx1250)
__device__ __forceinline__ unsigned short f2bf(float f) {
    union { __bf16 h; unsigned short s; } u;
    u.h = (__bf16)f;
    return u.s;
}

// packed pair conversion: lets clang pick the packed cvt form if available
__device__ __forceinline__ unsigned pk2(float a, float b) {
    v2f f = {a, b};
    union { v2bf h; unsigned u; } u;
    u.h = __builtin_convertvector(f, v2bf);
    return u.u;
}

__device__ __forceinline__ float fast_sigmoid(float v) {
    return 1.0f / (1.0f + __expf(-v));
}

// epilogue modes
//  0: C = acc
//  1: C = lrelu(acc + bias[n], 0.2)
//  2: C = acc + bias[n]
//  3: C = sigmoid(acc)
//  4: C += sigmoid(acc)
template<int EPI>
__device__ __forceinline__ void store_epi(float* __restrict__ C,
                                          const float* __restrict__ bias,
                                          int m, int n, int N, float v) {
    size_t off = (size_t)m * (size_t)N + (size_t)n;
    if (EPI == 1) {
        v += bias[n];
        C[off] = v > 0.0f ? v : 0.2f * v;
    } else if (EPI == 2) {
        C[off] = v + bias[n];
    } else if (EPI == 3) {
        C[off] = fast_sigmoid(v);
    } else if (EPI == 4) {
        C[off] += fast_sigmoid(v);
    } else {
        C[off] = v;
    }
}

// ---------------------------------------------------------------------------
// Tiled bf16 WMMA GEMM: C[M,N] = epi(A[M,K] @ op(B))
//   TRANSB=false : B is [K,N] row-major
//   TRANSB=true  : B is [N,K] row-major (C = A @ B^T)
// Block: 256 threads = 8 waves (4 x 2). Tile 128x64, K-step 32.
// Each wave computes a 32x32 slab with four v_wmma_f32_16x16x32_bf16.
// Double-buffered LDS + register staging: global loads for tile i+1 are in
// flight while WMMAs run on tile i; one barrier per K-step.
// REQUIRES: K % 4 == 0 and N % 4 == 0 (true for every GEMM in this model) —
// guards are branchless clamp+select on loaded values, never exec branches.
// ---------------------------------------------------------------------------
template<int EPI, bool TRANSB>
__global__ __launch_bounds__(256) void gemm_wmma(const float* __restrict__ A,
                                                 const float* __restrict__ Bm,
                                                 const float* __restrict__ bias,
                                                 float* __restrict__ C,
                                                 int M, int N, int K) {
    // 80-byte row stride (40 shorts) -> 16B aligned b128 slots, no bank conflicts
    __shared__ __align__(16) unsigned short As[2][128][40];
    __shared__ __align__(16) unsigned short Bs[2][64][40];   // stored as [n][k]

    const int t    = threadIdx.x;
    const int lane = t & 31;
    const int wave = t >> 5;
    const int wm   = wave & 3;     // 0..3  (M direction, 32 rows each)
    const int wn   = wave >> 2;    // 0..1  (N direction, 32 cols each)
    const int m0   = blockIdx.y * 128;
    const int n0   = blockIdx.x * 64;

    const int lm = lane & 15;
    const int hi = lane >> 4;

    v8f acc00 = {}, acc01 = {}, acc10 = {}, acc11 = {};

    float4 ra[4];   // staged A values for the in-flight K tile
    float4 rb[2];   // staged B values for the in-flight K tile

    // issue global loads for K-tile at k0 into registers (no wait here);
    // selects are applied to the loaded floats only -> v_cndmask, no branches
    auto loadGlobal = [&](int k0) {
#pragma unroll
        for (int i = 0; i < 4; ++i) {
            int idx = t + i * 256;         // 0..1023
            int r   = idx >> 3;            // 0..127
            int c4  = (idx & 7) << 2;      // 0,4,..,28
            int gm  = m0 + r;
            int gk  = k0 + c4;
            bool ok = (gm < M) && (gk < K);
            const float* p = A + (size_t)(gm < M ? gm : 0) * K + (gk < K ? gk : 0);
            float4 v = *(const float4*)p;
            ra[i].x = ok ? v.x : 0.0f;
            ra[i].y = ok ? v.y : 0.0f;
            ra[i].z = ok ? v.z : 0.0f;
            ra[i].w = ok ? v.w : 0.0f;
        }
        if (TRANSB) {
            // Bm is [N,K]: rows map directly to Bs rows -> vectorized along K
#pragma unroll
            for (int i = 0; i < 2; ++i) {
                int idx = t + i * 256;     // 0..511
                int n   = idx >> 3;        // 0..63
                int c4  = (idx & 7) << 2;  // 0..28
                int gn  = n0 + n;
                int gk  = k0 + c4;
                bool ok = (gn < N) && (gk < K);
                const float* p = Bm + (size_t)(gn < N ? gn : 0) * K + (gk < K ? gk : 0);
                float4 v = *(const float4*)p;
                rb[i].x = ok ? v.x : 0.0f;
                rb[i].y = ok ? v.y : 0.0f;
                rb[i].z = ok ? v.z : 0.0f;
                rb[i].w = ok ? v.w : 0.0f;
            }
        } else {
            // Bm is [K,N]: vectorized global read along N
#pragma unroll
            for (int i = 0; i < 2; ++i) {
                int idx = t + i * 256;     // 0..511
                int kk  = idx >> 4;        // 0..31
                int n4  = (idx & 15) << 2; // 0..60
                int gk  = k0 + kk;
                int gn  = n0 + n4;
                bool ok = (gk < K) && (gn < N);
                const float* p = Bm + (size_t)(gk < K ? gk : 0) * N + (gn < N ? gn : 0);
                float4 v = *(const float4*)p;
                rb[i].x = ok ? v.x : 0.0f;
                rb[i].y = ok ? v.y : 0.0f;
                rb[i].z = ok ? v.z : 0.0f;
                rb[i].w = ok ? v.w : 0.0f;
            }
        }
    };

    // convert staged registers to bf16 and write LDS buffer pb
    auto storeLds = [&](int pb) {
#pragma unroll
        for (int i = 0; i < 4; ++i) {
            int idx = t + i * 256;
            int r   = idx >> 3;
            int c4  = (idx & 7) << 2;
            uint2 p;
            p.x = pk2(ra[i].x, ra[i].y);
            p.y = pk2(ra[i].z, ra[i].w);
            *(uint2*)&As[pb][r][c4] = p;
        }
        if (TRANSB) {
#pragma unroll
            for (int i = 0; i < 2; ++i) {
                int idx = t + i * 256;
                int n   = idx >> 3;
                int c4  = (idx & 7) << 2;
                uint2 p;
                p.x = pk2(rb[i].x, rb[i].y);
                p.y = pk2(rb[i].z, rb[i].w);
                *(uint2*)&Bs[pb][n][c4] = p;
            }
        } else {
#pragma unroll
            for (int i = 0; i < 2; ++i) {
                int idx = t + i * 256;
                int kk  = idx >> 4;
                int n4  = (idx & 15) << 2;
                Bs[pb][n4 + 0][kk] = f2bf(rb[i].x);
                Bs[pb][n4 + 1][kk] = f2bf(rb[i].y);
                Bs[pb][n4 + 2][kk] = f2bf(rb[i].z);
                Bs[pb][n4 + 3][kk] = f2bf(rb[i].w);
            }
        }
    };

    loadGlobal(0);
    int pb = 0;
    for (int k0 = 0; k0 < K; k0 += 32) {
        storeLds(pb);
        __syncthreads();
        if (k0 + 32 < K) loadGlobal(k0 + 32);   // overlap with WMMA below

        // ---- A fragments: row = wm*32 + {0,16} + lm; K packed {hi*8.., 16+hi*8..}
        Frag fa0, fa1, fb0, fb1;
        const int ar = wm * 32 + lm;
        fa0.q[0] = *(const uint4*)&As[pb][ar][hi * 8];
        fa0.q[1] = *(const uint4*)&As[pb][ar][16 + hi * 8];
        fa1.q[0] = *(const uint4*)&As[pb][ar + 16][hi * 8];
        fa1.q[1] = *(const uint4*)&As[pb][ar + 16][16 + hi * 8];

        // ---- B fragments: col = wn*32 + {0,16} + lm; K packed {hi*8.., 16+hi*8..}
        const int bc = wn * 32 + lm;
        fb0.q[0] = *(const uint4*)&Bs[pb][bc][hi * 8];
        fb0.q[1] = *(const uint4*)&Bs[pb][bc][16 + hi * 8];
        fb1.q[0] = *(const uint4*)&Bs[pb][bc + 16][hi * 8];
        fb1.q[1] = *(const uint4*)&Bs[pb][bc + 16][16 + hi * 8];

        acc00 = __builtin_amdgcn_wmma_f32_16x16x32_bf16(
            false, fa0.v, false, fb0.v, (short)0, acc00, false, false);
        acc01 = __builtin_amdgcn_wmma_f32_16x16x32_bf16(
            false, fa0.v, false, fb1.v, (short)0, acc01, false, false);
        acc10 = __builtin_amdgcn_wmma_f32_16x16x32_bf16(
            false, fa1.v, false, fb0.v, (short)0, acc10, false, false);
        acc11 = __builtin_amdgcn_wmma_f32_16x16x32_bf16(
            false, fa1.v, false, fb1.v, (short)0, acc11, false, false);
        pb ^= 1;
    }

    // ---- epilogue: lane layout VGPR r -> M = hi*8 + r, N = lm
    const int gmb = m0 + wm * 32 + hi * 8;
    const int gnb = n0 + wn * 32 + lm;
#pragma unroll
    for (int r = 0; r < 8; ++r) {
        int gm0 = gmb + r;
        int gm1 = gm0 + 16;
        if (gm0 < M) {
            if (gnb < N)      store_epi<EPI>(C, bias, gm0, gnb,      N, acc00[r]);
            if (gnb + 16 < N) store_epi<EPI>(C, bias, gm0, gnb + 16, N, acc01[r]);
        }
        if (gm1 < M) {
            if (gnb < N)      store_epi<EPI>(C, bias, gm1, gnb,      N, acc10[r]);
            if (gnb + 16 < N) store_epi<EPI>(C, bias, gm1, gnb + 16, N, acc11[r]);
        }
    }
}

// ---------------------------------------------------------------------------
// z_i = a * z_ae + (1 - a) * z_sgae          (elementwise, a is [N,NZ])
// ---------------------------------------------------------------------------
__global__ void mix_kernel(const float* __restrict__ a,
                           const float* __restrict__ zae,
                           const float* __restrict__ zsg,
                           float* __restrict__ zi, int n) {
    int i = blockIdx.x * 256 + threadIdx.x;
    if (i < n) {
        float av = a[i];
        zi[i] = av * zae[i] + (1.0f - av) * zsg[i];
    }
}

// z_tilde = gamma * z_g + z_l   (gamma is a 1-element device array)
__global__ void gamma_axpy_kernel(const float* __restrict__ gamma,
                                  const float* __restrict__ zg,
                                  const float* __restrict__ zl,
                                  float* __restrict__ zt, int n) {
    int i = blockIdx.x * 256 + threadIdx.x;
    if (i < n) zt[i] = gamma[0] * zg[i] + zl[i];
}

// ---------------------------------------------------------------------------
// Flash-style row softmax attention:  z_g = softmax(z_l z_l^T) @ z_l, NZ=20
// One thread per row; z_l streamed through LDS in 256-row tiles.
// ---------------------------------------------------------------------------
__global__ __launch_bounds__(256) void attn_kernel(const float* __restrict__ zl,
                                                   float* __restrict__ zg,
                                                   int Nrow) {
    __shared__ float tile[256][20];
    const int i = blockIdx.x * 256 + threadIdx.x;

    float zi[20];
#pragma unroll
    for (int k = 0; k < 20; ++k) zi[k] = (i < Nrow) ? zl[(size_t)i * 20 + k] : 0.0f;

    float mmax = -3.4e38f, lsum = 0.0f;
    float acc[20];
#pragma unroll
    for (int k = 0; k < 20; ++k) acc[k] = 0.0f;

    for (int j0 = 0; j0 < Nrow; j0 += 256) {
        for (int idx = threadIdx.x; idx < 256 * 20; idx += 256) {
            int jj = idx / 20, kk = idx % 20;
            tile[jj][kk] = zl[(size_t)(j0 + jj) * 20 + kk];
        }
        __syncthreads();
        for (int j = 0; j < 256; ++j) {
            float s = 0.0f;
#pragma unroll
            for (int k = 0; k < 20; ++k) s += zi[k] * tile[j][k];
            float nm   = fmaxf(mmax, s);
            float corr = __expf(mmax - nm);
            float p    = __expf(s - nm);
            lsum = lsum * corr + p;
#pragma unroll
            for (int k = 0; k < 20; ++k) acc[k] = acc[k] * corr + p * tile[j][k];
            mmax = nm;
        }
        __syncthreads();
    }
    if (i < Nrow) {
        float inv = 1.0f / lsum;
#pragma unroll
        for (int k = 0; k < 20; ++k) zg[(size_t)i * 20 + k] = acc[k] * inv;
    }
}

// ---------------------------------------------------------------------------
// Student-t soft assignment (V=1):  q_ij = (1+||z_i-c_j||^2)^-1, row-normalized
// ---------------------------------------------------------------------------
__global__ __launch_bounds__(256) void soft_assign_kernel(const float* __restrict__ z,
                                                          const float* __restrict__ cl,
                                                          float* __restrict__ q,
                                                          int Nrow) {
    int i = blockIdx.x * 256 + threadIdx.x;
    if (i >= Nrow) return;
    float zi[20];
#pragma unroll
    for (int k = 0; k < 20; ++k) zi[k] = z[(size_t)i * 20 + k];
    float qv[10], sum = 0.0f;
#pragma unroll
    for (int j = 0; j < 10; ++j) {
        float d2 = 0.0f;
#pragma unroll
        for (int k = 0; k < 20; ++k) {
            float d = zi[k] - cl[j * 20 + k];
            d2 += d * d;
        }
        float t = 1.0f / (1.0f + d2);
        qv[j] = t;
        sum += t;
    }
    float inv = 1.0f / sum;
#pragma unroll
    for (int j = 0; j < 10; ++j) q[(size_t)i * 10 + j] = qv[j] * inv;
}

// ---------------------------------------------------------------------------
// host-side dispatch helper
// ---------------------------------------------------------------------------
static void gemm(hipStream_t s, const float* A, const float* B, const float* bias,
                 float* C, int M, int N, int K, bool transB, int epi) {
    dim3 grid((N + 63) / 64, (M + 127) / 128);
    dim3 blk(256);
    if (!transB) {
        if (epi == 0)      gemm_wmma<0, false><<<grid, blk, 0, s>>>(A, B, bias, C, M, N, K);
        else if (epi == 1) gemm_wmma<1, false><<<grid, blk, 0, s>>>(A, B, bias, C, M, N, K);
        else               gemm_wmma<2, false><<<grid, blk, 0, s>>>(A, B, bias, C, M, N, K);
    } else {
        if (epi == 3)      gemm_wmma<3, true><<<grid, blk, 0, s>>>(A, B, bias, C, M, N, K);
        else               gemm_wmma<4, true><<<grid, blk, 0, s>>>(A, B, bias, C, M, N, K);
    }
}

extern "C" void kernel_launch(void* const* d_in, const int* in_sizes, int n_in,
                              void* d_out, int out_size, void* d_ws, size_t ws_size,
                              hipStream_t stream) {
    (void)in_sizes; (void)n_in; (void)out_size; (void)ws_size;

    const float* x       = (const float*)d_in[0];
    const float* adj     = (const float*)d_in[1];
    const float* a       = (const float*)d_in[2];
    const float* cluster = (const float*)d_in[3];
    const float* gamma   = (const float*)d_in[4];
    const float* We1 = (const float*)d_in[5];  const float* be1 = (const float*)d_in[6];
    const float* We2 = (const float*)d_in[7];  const float* be2 = (const float*)d_in[8];
    const float* We3 = (const float*)d_in[9];  const float* be3 = (const float*)d_in[10];
    const float* Wz  = (const float*)d_in[11]; const float* bz  = (const float*)d_in[12];
    const float* Wd1 = (const float*)d_in[13]; const float* bd1 = (const float*)d_in[14];
    const float* Wd2 = (const float*)d_in[15]; const float* bd2 = (const float*)d_in[16];
    const float* Wd3 = (const float*)d_in[17]; const float* bd3 = (const float*)d_in[18];
    const float* Wx  = (const float*)d_in[19]; const float* bx  = (const float*)d_in[20];
    const float* Wg1 = (const float*)d_in[21];
    const float* Wg2 = (const float*)d_in[22];
    const float* Wg3 = (const float*)d_in[23];
    const float* Wg4 = (const float*)d_in[24];
    const float* Wg5 = (const float*)d_in[25];
    const float* Wg6 = (const float*)d_in[26];

    const size_t Nn = 8192, Dd = 2000, Zz = 20, Cc = 10;

    // outputs in tuple return order
    float* out      = (float*)d_out;
    float* o_xhat   = out;
    float* o_zhat   = o_xhat + Nn * Dd;
    float* o_adjhat = o_zhat + Nn * Dd;
    float* o_zae    = o_adjhat + Nn * Nn;
    float* o_zsgae  = o_zae + Nn * Zz;
    float* o_q      = o_zsgae + Nn * Zz;
    float* o_q1     = o_q + Nn * Cc;
    float* o_q2     = o_q1 + Nn * Cc;
    float* o_zt     = o_q2 + Nn * Cc;

    // workspace layout (floats)
    float* ws     = (float*)d_ws;
    float* bufA   = ws;                    // 8192 x 512
    float* bufB   = bufA + Nn * 512;       // 8192 x 512
    float* bufBig = bufB + Nn * 512;       // 8192 x 2000
    float* w_zi   = bufBig + Nn * Dd;      // 8192 x 20
    float* w_zl   = w_zi + Nn * Zz;        // 8192 x 20
    float* w_zg   = w_zl + Nn * Zz;        // 8192 x 20

    // ---- AE encoder ----
    gemm(stream, x,    We1, be1, bufA,  8192, 128,  2000, false, 1);
    gemm(stream, bufA, We2, be2, bufB,  8192, 256,  128,  false, 1);
    gemm(stream, bufB, We3, be3, bufA,  8192, 512,  256,  false, 1);
    gemm(stream, bufA, Wz,  bz,  o_zae, 8192, 20,   512,  false, 2);

    // ---- SGAE encoder (no activation in GNN layers) ----
    gemm(stream, x,    Wg1,  nullptr, bufA,    8192, 128, 2000, false, 0);
    gemm(stream, adj,  bufA, nullptr, bufB,    8192, 128, 8192, false, 0);
    gemm(stream, bufB, Wg2,  nullptr, bufA,    8192, 256, 128,  false, 0);
    gemm(stream, adj,  bufA, nullptr, bufB,    8192, 256, 8192, false, 0);
    gemm(stream, bufB, Wg3,  nullptr, bufA,    8192, 20,  256,  false, 0);
    gemm(stream, adj,  bufA, nullptr, o_zsgae, 8192, 20,  8192, false, 0);

    // ---- fuse + local/global attention ----
    {
        int n = (int)(Nn * Zz);
        mix_kernel<<<(n + 255) / 256, 256, 0, stream>>>(a, o_zae, o_zsgae, w_zi, n);
        gemm(stream, adj, w_zi, nullptr, w_zl, 8192, 20, 8192, false, 0);
        attn_kernel<<<8192 / 256, 256, 0, stream>>>(w_zl, w_zg, 8192);
        gamma_axpy_kernel<<<(n + 255) / 256, 256, 0, stream>>>(gamma, w_zg, w_zl, o_zt, n);
    }

    // ---- AE decoder ----
    gemm(stream, o_zt, Wd1, bd1, bufA,   8192, 512,  20,  false, 1);
    gemm(stream, bufA, Wd2, bd2, bufB,   8192, 256,  512, false, 1);
    gemm(stream, bufB, Wd3, bd3, bufA,   8192, 128,  256, false, 1);
    gemm(stream, bufA, Wx,  bx,  o_xhat, 8192, 2000, 128, false, 2);

    // ---- SGAE decoder ----
    gemm(stream, o_zt, Wg4,    nullptr, bufA,   8192, 256,  20,   false, 0);
    gemm(stream, adj,  bufA,   nullptr, bufB,   8192, 256,  8192, false, 0);
    gemm(stream, bufB, Wg5,    nullptr, bufA,   8192, 128,  256,  false, 0);
    gemm(stream, adj,  bufA,   nullptr, bufB,   8192, 128,  8192, false, 0);
    gemm(stream, bufB, Wg6,    nullptr, bufBig, 8192, 2000, 128,  false, 0);
    gemm(stream, adj,  bufBig, nullptr, o_zhat, 8192, 2000, 8192, false, 0);

    // ---- adj_hat = sigmoid(z_sgae z_sgae^T) + sigmoid(z_hat z_hat^T) ----
    // fused epilogues avoid any N x N scratch buffer
    gemm(stream, o_zhat,  o_zhat,  nullptr, o_adjhat, 8192, 8192, 2000, true, 3);
    gemm(stream, o_zsgae, o_zsgae, nullptr, o_adjhat, 8192, 8192, 20,   true, 4);

    // ---- student-t soft assignments (V = 1) ----
    soft_assign_kernel<<<8192 / 256, 256, 0, stream>>>(o_zt,    cluster, o_q,  8192);
    soft_assign_kernel<<<8192 / 256, 256, 0, stream>>>(o_zae,   cluster, o_q1, 8192);
    soft_assign_kernel<<<8192 / 256, 256, 0, stream>>>(o_zsgae, cluster, o_q2, 8192);
}
